// LSTM_13159779795398
// MI455X (gfx1250) — matrix operand order repaired and checked by reading, not compiled, not run on hardware
//
#include <hip/hip_runtime.h>
#include <hip/hip_bf16.h>
#include <math.h>

typedef __bf16 bf16;
typedef __attribute__((ext_vector_type(16))) __bf16 v16bf;
typedef __attribute__((ext_vector_type(8)))  __bf16 v8bf;
typedef __attribute__((ext_vector_type(8)))  float  v8f;

#define T_STEPS 4096
#define ISZ 512
#define HSZ 2048
#define G4  8192            // 4*HSZ
#define NBLK (HSZ / 16)     // 128 persistent blocks
#define LSTRIDE 2056        // padded LDS row stride (elems): bank-conflict-free b128 reads

// ---------------- prep kernels ----------------

__global__ void k_cvt(const float* __restrict__ s, bf16* __restrict__ d, int n) {
    int i = blockIdx.x * blockDim.x + threadIdx.x;
    if (i < n) d[i] = (bf16)s[i];
}

__global__ void k_bias(const float* __restrict__ a, const float* __restrict__ b,
                       float* __restrict__ o, int n) {
    int i = blockIdx.x * blockDim.x + threadIdx.x;
    if (i < n) o[i] = a[i] + b[i];
}

__global__ void k_init(bf16* __restrict__ h0, bf16* __restrict__ h1,
                       unsigned int* __restrict__ bar) {
    int i = blockIdx.x * blockDim.x + threadIdx.x;
    if (i < HSZ) { h0[i] = (bf16)0.0f; h1[i] = (bf16)0.0f; }
    if (i == 0) *bar = 0u;
}

// ---------------- phase 1: x_gates = X @ W_ih^T + bias  (bf16 WMMA) ----------------
// One wave per 16x16 output tile; K loop over 512 in chunks of 32.
// Output tile staged through LDS so stores are coalesced b128 (1 per lane)
// instead of 8 scattered b16 per lane.
__global__ __launch_bounds__(128)
void k_gemm_xg(const bf16* __restrict__ X, const bf16* __restrict__ Wih,
               const float* __restrict__ bias, bf16* __restrict__ xg) {
    __shared__ bf16 stg[4][16][16];                  // per-wave 16x16 bf16 tile
    const int lane = threadIdx.x & 31;
    const int wave = threadIdx.x >> 5;
    const int tile = blockIdx.x * 4 + wave;          // 131072 tiles total
    const int tilesG = G4 / 16;                      // 512
    const int tt = tile / tilesG;
    const int gg = tile % tilesG;
    const int t0 = tt * 16, g0 = gg * 16;
    const int m  = lane & 15;
    const bool hi = lane >= 16;

    const v8bf* pa = (const v8bf*)(X   + (size_t)(t0 + m) * ISZ);
    const v8bf* pb = (const v8bf*)(Wih + (size_t)(g0 + m) * ISZ);

    v8f acc = {};
    #pragma unroll 4
    for (int kk = 0; kk < ISZ / 32; ++kk) {
        const int k0 = kk * 32;
        const int ka = k0 + (hi ? 8 : 0);
        const int kb = k0 + (hi ? 8 : 0);
        union { v16bf v; v8bf h[2]; } A, B;
        A.h[0] = pa[ka >> 3];
        A.h[1] = pa[(ka + 16) >> 3];
        B.h[0] = pb[kb >> 3];
        B.h[1] = pb[(kb >> 3) + 2];
        acc = __builtin_amdgcn_wmma_f32_16x16x32_bf16(
            false, A.v, false, B.v, (short)0, acc, false, false);
    }
    const float bv = bias[g0 + m];
    #pragma unroll
    for (int v = 0; v < 8; ++v)
        stg[wave][v + (hi ? 8 : 0)][m] = (bf16)(acc[v] + bv);
    __syncthreads();

    // lane l writes half a row: row r = l>>1, 8 elems at (l&1)*8 -> one b128 store
    const int r = lane >> 1, half = lane & 1;
    v8bf val = *(const v8bf*)&stg[wave][r][half * 8];
    *(v8bf*)(xg + (size_t)(t0 + r) * G4 + g0 + half * 8) = val;
}

// ---------------- phase 2: persistent LSTM recurrence ----------------
// 128 blocks x 128 threads; block b owns hidden units j0..j0+15 for the whole
// sequence (cell state in registers). Gates i,f weights (128.5 KB) live in LDS
// for all 4096 steps (loaded once); gates g,o stream from L2. 4 waves K-split
// 2048 -> 4x512; WMMA with B = h broadcast; LDS reduce; device-wide monotonic
// barrier per step (release wb -> arrive -> spin/s_sleep -> acquire inv).
// ~129.5 KB LDS/block => 2 blocks/WGP (<=320 KB), barrier-safe on >=64 WGPs.
__global__ __launch_bounds__(128)
void k_lstm_persist(const bf16* __restrict__ Whh, const bf16* __restrict__ xg,
                    bf16* __restrict__ h0, bf16* __restrict__ h1,
                    unsigned int* __restrict__ bar) {
    extern __shared__ __align__(16) char smem_raw[];
    bf16* wlds = (bf16*)smem_raw;          // [2 gates][16 rows][LSTRIDE]
    __shared__ float part[4][4][16];       // [wave][gate][row]

    const int tid  = threadIdx.x;
    const int lane = tid & 31;
    const int wave = tid >> 5;
    const int j0 = blockIdx.x * 16;
    const int m  = lane & 15;
    const bool hi = lane >= 16;
    const int kw0 = wave * (HSZ / 4);      // 512 per wave

    // ---- one-time copy of gate-i / gate-f weight rows into LDS ----
    for (int idx = tid; idx < 2 * 16 * (HSZ / 8); idx += 128) {
        const int q  = idx / (16 * (HSZ / 8));
        const int rm = idx % (16 * (HSZ / 8));
        const int r  = rm / (HSZ / 8);
        const int kc = rm % (HSZ / 8);
        const v8bf* src = (const v8bf*)(Whh + (size_t)(q * HSZ + j0 + r) * HSZ) + kc;
        *((v8bf*)(wlds + (size_t)(q * 16 + r) * LSTRIDE) + kc) = *src;
    }
    __syncthreads();

    // gates g,o stream from global (L2-resident across steps)
    const v8bf* pa2 = (const v8bf*)(Whh + (size_t)(2 * HSZ + j0 + m) * HSZ);
    const v8bf* pa3 = (const v8bf*)(Whh + (size_t)(3 * HSZ + j0 + m) * HSZ);
    const v8bf* la0 = (const v8bf*)(wlds + (size_t)(0 * 16 + m) * LSTRIDE);
    const v8bf* la1 = (const v8bf*)(wlds + (size_t)(1 * 16 + m) * LSTRIDE);

    float cj = 0.0f;                       // cell state, register-resident (tid<16)

    for (int t = 0; t < T_STEPS; ++t) {
        // prefetch next step's x_gates off the serial critical path (HBM stream)
        if (tid < 16 && t + 1 < T_STEPS) {
            const bf16* nx = xg + (size_t)(t + 1) * G4 + j0 + tid;
            __builtin_prefetch(nx, 0, 1);
            __builtin_prefetch(nx + HSZ, 0, 1);
            __builtin_prefetch(nx + 2 * HSZ, 0, 1);
            __builtin_prefetch(nx + 3 * HSZ, 0, 1);
        }

        const bf16* h_in  = (t & 1) ? h1 : h0;
        bf16*       h_out = (t & 1) ? h0 : h1;
        const v8bf* pb = (const v8bf*)h_in;

        v8f acc0 = {}, acc1 = {}, acc2 = {}, acc3 = {};
        #pragma unroll 4
        for (int kk = 0; kk < 16; ++kk) {
            const int k0 = kw0 + kk * 32;
            const int ka = k0 + (hi ? 8 : 0);
            const int kb = k0 + (hi ? 8 : 0);
            union { v16bf v; v8bf h[2]; } A, B;
            B.h[0] = pb[kb >> 3];
            B.h[1] = pb[(kb >> 3) + 2];

            A.h[0] = la0[ka >> 3]; A.h[1] = la0[(ka >> 3) + 2];   // ds_load_b128
            acc0 = __builtin_amdgcn_wmma_f32_16x16x32_bf16(false, A.v, false, B.v, (short)0, acc0, false, false);
            A.h[0] = la1[ka >> 3]; A.h[1] = la1[(ka >> 3) + 2];
            acc1 = __builtin_amdgcn_wmma_f32_16x16x32_bf16(false, A.v, false, B.v, (short)0, acc1, false, false);
            A.h[0] = pa2[ka >> 3]; A.h[1] = pa2[(ka >> 3) + 2];   // global b128 (L2)
            acc2 = __builtin_amdgcn_wmma_f32_16x16x32_bf16(false, A.v, false, B.v, (short)0, acc2, false, false);
            A.h[0] = pa3[ka >> 3]; A.h[1] = pa3[(ka >> 3) + 2];
            acc3 = __builtin_amdgcn_wmma_f32_16x16x32_bf16(false, A.v, false, B.v, (short)0, acc3, false, false);
        }

        // B columns identical (broadcast) -> D column 0 holds y; lanes 0 & 16 own it.
        if (m == 0) {
            const int rb = hi ? 8 : 0;
            #pragma unroll
            for (int v = 0; v < 8; ++v) {
                part[wave][0][rb + v] = acc0[v];
                part[wave][1][rb + v] = acc1[v];
                part[wave][2][rb + v] = acc2[v];
                part[wave][3][rb + v] = acc3[v];
            }
        }
        __syncthreads();

        if (tid < 16) {
            const int j = j0 + tid;
            const bf16* xg_t = xg + (size_t)t * G4;
            float s[4];
            #pragma unroll
            for (int q = 0; q < 4; ++q) {
                float v = (float)xg_t[q * HSZ + j];
                v += part[0][q][tid] + part[1][q][tid]
                   + part[2][q][tid] + part[3][q][tid];
                s[q] = v;
            }
            const float ig = 1.0f / (1.0f + __expf(-s[0]));
            const float fg = 1.0f / (1.0f + __expf(-s[1]));
            const float gg = tanhf(s[2]);
            const float og = 1.0f / (1.0f + __expf(-s[3]));
            cj = fg * cj + ig * gg;
            h_out[j] = (bf16)(og * tanhf(cj));
        }
        __syncthreads();

        // ---- device-wide barrier (monotonic arrive counter) ----
        if (tid == 0) {
            __builtin_amdgcn_fence(__ATOMIC_RELEASE, "agent");   // flush h stores to L2
            __hip_atomic_fetch_add(bar, 1u, __ATOMIC_RELAXED, __HIP_MEMORY_SCOPE_AGENT);
            const unsigned int target = (unsigned int)(t + 1) * (unsigned int)NBLK;
            while (__hip_atomic_load(bar, __ATOMIC_RELAXED, __HIP_MEMORY_SCOPE_AGENT) < target) {
                __builtin_amdgcn_s_sleep(2);
            }
        }
        __syncthreads();
        // every wave acquires (both CU memory paths of the WGP invalidate stale h)
        __builtin_amdgcn_fence(__ATOMIC_ACQUIRE, "agent");
    }
}

// ---------------- phase 3: out = h_T . W_lin + b_lin ----------------
__global__ void k_final(const bf16* __restrict__ hT, const float* __restrict__ Wlin,
                        const float* __restrict__ blin, float* __restrict__ out) {
    __shared__ float red[256];
    float s = 0.0f;
    for (int i = threadIdx.x; i < HSZ; i += 256) s += (float)hT[i] * Wlin[i];
    red[threadIdx.x] = s;
    __syncthreads();
    for (int off = 128; off > 0; off >>= 1) {
        if (threadIdx.x < off) red[threadIdx.x] += red[threadIdx.x + off];
        __syncthreads();
    }
    if (threadIdx.x == 0) out[0] = red[0] + blin[0];
}

// ---------------- launcher ----------------
extern "C" void kernel_launch(void* const* d_in, const int* in_sizes, int n_in,
                              void* d_out, int out_size, void* d_ws, size_t ws_size,
                              hipStream_t stream) {
    (void)in_sizes; (void)n_in; (void)out_size; (void)ws_size;
    const float* x    = (const float*)d_in[0];
    const float* Wih  = (const float*)d_in[1];
    const float* Whh  = (const float*)d_in[2];
    const float* bih  = (const float*)d_in[3];
    const float* bhh  = (const float*)d_in[4];
    const float* Wlin = (const float*)d_in[5];
    const float* blin = (const float*)d_in[6];
    float* out = (float*)d_out;
    char* ws = (char*)d_ws;

    // workspace layout (all offsets 4 KB aligned)
    bf16*  xg   = (bf16*)(ws);                       // 4096*8192*2  = 64 MB
    bf16*  WihB = (bf16*)(ws + 67108864ULL);         // 8192*512*2   = 8 MB
    bf16*  WhhB = (bf16*)(ws + 75497472ULL);         // 8192*2048*2  = 32 MB
    bf16*  xB   = (bf16*)(ws + 109051904ULL);        // 4096*512*2   = 4 MB
    float* bias = (float*)(ws + 113246208ULL);       // 8192*4
    bf16*  h0   = (bf16*)(ws + 113278976ULL);        // 2048*2
    bf16*  h1   = (bf16*)(ws + 113283072ULL);        // 2048*2
    unsigned int* bar = (unsigned int*)(ws + 113287168ULL);

    k_cvt<<<(G4 * ISZ + 255) / 256, 256, 0, stream>>>(Wih, WihB, G4 * ISZ);
    k_cvt<<<(G4 * HSZ + 255) / 256, 256, 0, stream>>>(Whh, WhhB, G4 * HSZ);
    k_cvt<<<(T_STEPS * ISZ + 255) / 256, 256, 0, stream>>>(x, xB, T_STEPS * ISZ);
    k_bias<<<(G4 + 255) / 256, 256, 0, stream>>>(bih, bhh, bias, G4);
    k_init<<<(HSZ + 255) / 256, 256, 0, stream>>>(h0, h1, bar);

    // 256 x 512 = 131072 tiles of 16x16; 4 waves (tiles) per 128-thread block
    k_gemm_xg<<<131072 / 4, 128, 0, stream>>>(xB, WihB, bias, xg);

    // whole recurrence in ONE persistent kernel; gates i,f weights LDS-resident
    const int ldsBytes = 2 * 16 * LSTRIDE * (int)sizeof(bf16);   // 131584 B
    (void)hipFuncSetAttribute((const void*)k_lstm_persist,
                              hipFuncAttributeMaxDynamicSharedMemorySize, ldsBytes);
    k_lstm_persist<<<NBLK, 128, ldsBytes, stream>>>(WhhB, xg, h0, h1, bar);

    // t = 4095 is odd -> final h lands in h0
    k_final<<<1, 256, 0, stream>>>(h0, Wlin, blin, out);
}
